// TritonMoEMLP_58110907515445
// MI455X (gfx1250) — matrix operand order repaired and checked by reading, not compiled, not run on hardware
//
#include <hip/hip_runtime.h>
#include <hip/hip_bf16.h>

typedef __attribute__((ext_vector_type(16))) _Float16 v16h;
typedef __attribute__((ext_vector_type(8)))  _Float16 v8h;
typedef __attribute__((ext_vector_type(8)))  float    v8f;

#define T_TOK 2048
#define D_EMB 768
#define TW    12288
#define NEXP  16
#define TOPK  8

#define BM 128
#define BN 64
#define BK 64    // two WMMA K-steps per LDS stage
#define LDK 72   // LDS row stride in halves: 144 bytes, multiple of 16 -> aligned b128 frags

union FragH { v16h f; v8h h[2]; };

__device__ __forceinline__ float gelu_tanh(float v) {
  const float c0 = 0.7978845608028654f; // sqrt(2/pi)
  float v3 = v * v * v;
  return 0.5f * v * (1.0f + tanhf(c0 * (v + 0.044715f * v3)));
}

// gfx1250 async global->LDS copy (ASYNCcnt path), GVS addressing:
// mem = SGPR64 + VADDR_I32 + inst_offset ; LDS byte addr from VDST VGPR.
__device__ __forceinline__ void async_copy_b128(unsigned lds_byte_off,
                                                unsigned global_byte_off,
                                                const void* base) {
  asm volatile("global_load_async_to_lds_b128 %0, %1, %2"
               :
               : "v"(lds_byte_off), "v"(global_byte_off),
                 "s"((unsigned long long)(uintptr_t)base)
               : "memory");
}
__device__ __forceinline__ void wait_async0() {
  asm volatile("s_wait_asynccnt 0x0" ::: "memory");
}

// -------------------- 1) Router: logits -> softmax -> top-8 -> comb[T,16] ---
__global__ __launch_bounds__(128)
void moe_router_kernel(const float* __restrict__ x,
                       const float* __restrict__ rw,
                       float* __restrict__ comb) {
  const int wave = threadIdx.x >> 5;
  const int lane = threadIdx.x & 31;
  const int t = blockIdx.x * (blockDim.x >> 5) + wave;
  if (t >= T_TOK) return;

  float logit = 0.0f;
  if (lane < NEXP) {
    const float* xr = x + (size_t)t * D_EMB;
    const float* wr = rw + (size_t)lane * D_EMB;
    #pragma unroll 4
    for (int d = 0; d < D_EMB; ++d) logit += xr[d] * wr[d];
  }
  float p[NEXP];
  #pragma unroll
  for (int j = 0; j < NEXP; ++j) p[j] = __shfl(logit, j, 32);

  float mx = p[0];
  #pragma unroll
  for (int j = 1; j < NEXP; ++j) mx = fmaxf(mx, p[j]);
  float s = 0.0f;
  #pragma unroll
  for (int j = 0; j < NEXP; ++j) { p[j] = expf(p[j] - mx); s += p[j]; }
  const float inv = 1.0f / s;
  #pragma unroll
  for (int j = 0; j < NEXP; ++j) p[j] *= inv;

  if (lane < NEXP) {
    const float me = p[lane];
    int rank = 0;
    #pragma unroll
    for (int j = 0; j < NEXP; ++j)
      rank += ((p[j] > me) || (p[j] == me && j < lane)) ? 1 : 0;
    float ssel = 0.0f;
    #pragma unroll
    for (int j = 0; j < NEXP; ++j) {
      int rj = 0;
      #pragma unroll
      for (int k = 0; k < NEXP; ++k)
        rj += ((p[k] > p[j]) || (p[k] == p[j] && k < j)) ? 1 : 0;
      if (rj < TOPK) ssel += p[j];
    }
    comb[(size_t)t * NEXP + lane] = (rank < TOPK) ? (me / ssel) : 0.0f;
  }
}

// -------------------- 2) Up-projection GEMM + gelu + comb scale -> h (f16) --
__global__ __launch_bounds__(256)
void moe_gemm_up_kernel(const float* __restrict__ x,
                        const float* __restrict__ w1,
                        const float* __restrict__ comb,
                        _Float16* __restrict__ hbuf) {
  __shared__ alignas(16) _Float16 As[BM * LDK];
  __shared__ alignas(16) _Float16 Bs[BN * LDK];

  const int tid  = threadIdx.x;
  const int lane = tid & 31;
  const int wid  = tid >> 5;
  const int wm   = (wid & 3) << 5;   // 0,32,64,96
  const int wn   = (wid >> 2) << 5;  // 0,32
  const int m0   = blockIdx.y * BM;
  const int n0   = blockIdx.x * BN;

  v8f zero = {};
  v8f acc[2][2];
  for (int i = 0; i < 2; ++i) for (int j = 0; j < 2; ++j) acc[i][j] = zero;

  const int arow = lane & 15;
  const int aklo = (lane >> 4) * 8;
  const int bkl  = (lane >> 4) * 8;

  for (int k0 = 0; k0 < D_EMB; k0 += BK) {
    // stage A: x[m0..+127][k0..+63] f32 -> f16 LDS  (2048 float4 / block)
    #pragma unroll
    for (int p = 0; p < 8; ++p) {
      const int idx = p * 256 + tid;
      const int r = idx >> 4, c4 = idx & 15;
      float4 v = *reinterpret_cast<const float4*>(x + (size_t)(m0 + r) * D_EMB + k0 + c4 * 4);
      _Float16* dst = &As[r * LDK + c4 * 4];
      dst[0] = (_Float16)v.x; dst[1] = (_Float16)v.y;
      dst[2] = (_Float16)v.z; dst[3] = (_Float16)v.w;
    }
    // stage B transposed: w1[k0..+63][n0..+63] -> Bs[n][k]  (1024 float4 / block)
    #pragma unroll
    for (int p = 0; p < 4; ++p) {
      const int idx = p * 256 + tid;
      const int r = idx >> 4, c4 = idx & 15;   // r = k row 0..63
      float4 v = *reinterpret_cast<const float4*>(w1 + (size_t)(k0 + r) * TW + n0 + c4 * 4);
      const int nb = c4 * 4;
      Bs[(nb + 0) * LDK + r] = (_Float16)v.x;
      Bs[(nb + 1) * LDK + r] = (_Float16)v.y;
      Bs[(nb + 2) * LDK + r] = (_Float16)v.z;
      Bs[(nb + 3) * LDK + r] = (_Float16)v.w;
    }
    if (k0 + BK < D_EMB) {
      __builtin_prefetch(x + (size_t)(m0 + (tid >> 4)) * D_EMB + k0 + BK, 0, 1);
      __builtin_prefetch(w1 + (size_t)(k0 + BK + (tid >> 4)) * TW + n0, 0, 1);
    }
    __syncthreads();

    #pragma unroll
    for (int kk = 0; kk < BK; kk += 32) {
      FragH a[2], b[2];
      #pragma unroll
      for (int i = 0; i < 2; ++i) {
        const _Float16* ap = &As[(wm + i * 16 + arow) * LDK + kk];
        a[i].h[0] = *reinterpret_cast<const v8h*>(ap + aklo);
        a[i].h[1] = *reinterpret_cast<const v8h*>(ap + aklo + 16);
      }
      #pragma unroll
      for (int j = 0; j < 2; ++j) {
        const _Float16* bp = &Bs[(wn + j * 16 + (lane & 15)) * LDK + kk + bkl];
        b[j].h[0] = *reinterpret_cast<const v8h*>(bp);
        b[j].h[1] = *reinterpret_cast<const v8h*>(bp + 16);
      }
      #pragma unroll
      for (int i = 0; i < 2; ++i)
        #pragma unroll
        for (int j = 0; j < 2; ++j)
          acc[i][j] = __builtin_amdgcn_wmma_f32_16x16x32_f16(
              false, a[i].f, false, b[j].f, (short)0, acc[i][j], false, false);
    }
    __syncthreads();
  }

  // epilogue: gelu, scale by routing weight, store f16
  #pragma unroll
  for (int i = 0; i < 2; ++i)
    #pragma unroll
    for (int j = 0; j < 2; ++j) {
      const int col = n0 + wn + j * 16 + (lane & 15);
      const int e   = (col < 4096) ? (col >> 9) : (8 + ((col - 4096) >> 10));
      #pragma unroll
      for (int g = 0; g < 8; ++g) {
        const int row = m0 + wm + i * 16 + ((lane < 16) ? g : (g + 8));
        const float hv = gelu_tanh(acc[i][j][g]) * comb[(size_t)row * NEXP + e];
        hbuf[(size_t)row * TW + col] = (_Float16)hv;
      }
    }
}

// -------------------- 3) Down-projection GEMM: out = h @ w2 ----------------
__global__ __launch_bounds__(256)
void moe_gemm_down_kernel(const _Float16* __restrict__ hbuf,
                          const float* __restrict__ w2,
                          float* __restrict__ out) {
  __shared__ alignas(16) _Float16 As[BM * LDK];
  __shared__ alignas(16) _Float16 Bs[BN * LDK];

  const int tid  = threadIdx.x;
  const int lane = tid & 31;
  const int wid  = tid >> 5;
  const int wm   = (wid & 3) << 5;
  const int wn   = (wid >> 2) << 5;
  const int m0   = blockIdx.y * BM;
  const int n0   = blockIdx.x * BN;

  v8f zero = {};
  v8f acc[2][2];
  for (int i = 0; i < 2; ++i) for (int j = 0; j < 2; ++j) acc[i][j] = zero;

  const int arow = lane & 15;
  const int aklo = (lane >> 4) * 8;
  const int bkl  = (lane >> 4) * 8;

  // per-thread async A staging geometry: 128 rows x 8 b128-chunks = 1024 xfers
  const int a_r0 = tid >> 3;      // 0..31, +32 per pass
  const int a_c8 = tid & 7;       // b128 chunk (8 halves)

  for (int k0 = 0; k0 < TW; k0 += BK) {
    // stage A via gfx1250 async global->LDS (pure f16 copy, ASYNCcnt)
    #pragma unroll
    for (int p = 0; p < 4; ++p) {
      const int r = a_r0 + p * 32;
      const unsigned lds_off =
          (unsigned)(uintptr_t)&As[r * LDK + a_c8 * 8];
      const unsigned goff =
          (unsigned)(((size_t)(m0 + r) * TW + k0 + a_c8 * 8) * sizeof(_Float16));
      async_copy_b128(lds_off, goff, hbuf);
    }
    // stage B transposed: w2[k0..+63][n0..+63] f32 -> Bs[n][k] f16
    #pragma unroll
    for (int p = 0; p < 4; ++p) {
      const int idx = p * 256 + tid;
      const int r = idx >> 4, c4 = idx & 15;
      float4 v = *reinterpret_cast<const float4*>(w2 + (size_t)(k0 + r) * D_EMB + n0 + c4 * 4);
      const int nb = c4 * 4;
      Bs[(nb + 0) * LDK + r] = (_Float16)v.x;
      Bs[(nb + 1) * LDK + r] = (_Float16)v.y;
      Bs[(nb + 2) * LDK + r] = (_Float16)v.z;
      Bs[(nb + 3) * LDK + r] = (_Float16)v.w;
    }
    if (k0 + BK < TW) {
      __builtin_prefetch(hbuf + (size_t)(m0 + (tid >> 4)) * TW + k0 + BK, 0, 1);
      __builtin_prefetch(w2 + (size_t)(k0 + BK + (tid >> 4)) * D_EMB + n0, 0, 1);
    }
    wait_async0();        // this wave's async LDS writes are done
    __syncthreads();      // all waves' staging visible

    #pragma unroll
    for (int kk = 0; kk < BK; kk += 32) {
      FragH a[2], b[2];
      #pragma unroll
      for (int i = 0; i < 2; ++i) {
        const _Float16* ap = &As[(wm + i * 16 + arow) * LDK + kk];
        a[i].h[0] = *reinterpret_cast<const v8h*>(ap + aklo);
        a[i].h[1] = *reinterpret_cast<const v8h*>(ap + aklo + 16);
      }
      #pragma unroll
      for (int j = 0; j < 2; ++j) {
        const _Float16* bp = &Bs[(wn + j * 16 + (lane & 15)) * LDK + kk + bkl];
        b[j].h[0] = *reinterpret_cast<const v8h*>(bp);
        b[j].h[1] = *reinterpret_cast<const v8h*>(bp + 16);
      }
      #pragma unroll
      for (int i = 0; i < 2; ++i)
        #pragma unroll
        for (int j = 0; j < 2; ++j)
          acc[i][j] = __builtin_amdgcn_wmma_f32_16x16x32_f16(
              false, a[i].f, false, b[j].f, (short)0, acc[i][j], false, false);
    }
    __syncthreads();
  }

  #pragma unroll
  for (int i = 0; i < 2; ++i)
    #pragma unroll
    for (int j = 0; j < 2; ++j) {
      const int col = n0 + wn + j * 16 + (lane & 15);
      #pragma unroll
      for (int g = 0; g < 8; ++g) {
        const int row = m0 + wm + i * 16 + ((lane < 16) ? g : (g + 8));
        out[(size_t)row * D_EMB + col] = acc[i][j][g];
      }
    }
}

// ---------------------------------------------------------------------------
extern "C" void kernel_launch(void* const* d_in, const int* in_sizes, int n_in,
                              void* d_out, int out_size, void* d_ws, size_t ws_size,
                              hipStream_t stream) {
  (void)in_sizes; (void)n_in; (void)out_size; (void)ws_size;
  const float* x  = (const float*)d_in[0];   // [1,2048,768]
  const float* rw = (const float*)d_in[1];   // [16,768]
  const float* w1 = (const float*)d_in[2];   // [768,12288]
  const float* w2 = (const float*)d_in[3];   // [12288,768]
  float* out = (float*)d_out;                // [1,2048,768]

  float*    comb = (float*)d_ws;                                   // 2048*16 f32
  _Float16* hbuf = (_Float16*)((char*)d_ws + (size_t)T_TOK * NEXP * sizeof(float));

  moe_router_kernel<<<T_TOK / 4, 128, 0, stream>>>(x, rw, comb);
  moe_gemm_up_kernel<<<dim3(TW / BN, T_TOK / BM), 256, 0, stream>>>(x, w1, comb, hbuf);
  moe_gemm_down_kernel<<<dim3(D_EMB / BN, T_TOK / BM), 256, 0, stream>>>(hbuf, w2, out);
}
